// VectorQuantizerEMA_541165879922
// MI455X (gfx1250) — matrix operand order repaired and checked by reading, not compiled, not run on hardware
//
#include <hip/hip_runtime.h>
#include <math.h>

// ---------------- problem constants ----------------
#define NPOS   65536      // 16 * 64 * 64 spatial positions
#define DLAT   128        // latent / channel dim
#define KCODE  1024       // codebook entries
#define HW     4096       // 64*64
#define CIMG   524288     // 128 * 4096 floats per image (NCHW plane stride * C)

// output layout (flat, reference return order, all as fp32)
#define OUT_LOSS 0
#define OUT_Q    1
#define OUT_NEW  8388609      // 1 + 16*128*64*64
#define OUT_PERP 16777217     // 1 + 2*8388608
#define OUT_IDX  16777218

// ---------------- WMMA helpers ----------------
typedef __attribute__((ext_vector_type(16))) _Float16 v16h;
typedef __attribute__((ext_vector_type(8)))  float    v8f;

union ABfrag { v16h v; uint4 q[2]; };

__device__ __forceinline__ v8f wmma_f16(v16h a, v16h b, v8f c) {
  // D = A(16x32,f16) x B(32x16,f16) + C(16x16,f32)
  return __builtin_amdgcn_wmma_f32_16x16x32_f16(false, a, false, b,
                                                (short)0, c, false, false);
}

// A fragment from row-major [row][k] f16, k contiguous, row stride ld halves.
// Lane layout (ISA 7.12.2): m = lane&15; lanes<16 hold K {0..7,16..23}, lanes>=16 hold K {8..15,24..31}.
__device__ __forceinline__ v16h load_frag_A(const _Float16* base, int row0,
                                            int ld, int k0, int lane) {
  const int m = lane & 15, kh = lane >> 4;
  const _Float16* p = base + (size_t)(row0 + m) * ld + k0 + kh * 8;
  ABfrag f;
  f.q[0] = *(const uint4*)(p);        // K k0+kh*8 .. +7      -> VGPR 0-3
  f.q[1] = *(const uint4*)(p + 16);   // K k0+16+kh*8 .. +7   -> VGPR 4-7
  return f.v;
}

// B fragment from row-major [col][k] f16 (i.e. B stored transposed, K contiguous).
// Lane layout: n = lane&15; lanes<16 hold K {0..7,16..23}, lanes>=16 hold K {8..15,24..31}.
__device__ __forceinline__ v16h load_frag_B(const _Float16* base, int col0,
                                            int ld, int k0, int lane) {
  const int n = lane & 15, kh = lane >> 4;
  const _Float16* p = base + (size_t)(col0 + n) * ld + k0 + kh * 8;
  ABfrag f;
  f.q[0] = *(const uint4*)(p);        // K k0+kh*8 .. +7
  f.q[1] = *(const uint4*)(p + 16);   // K k0+16+kh*8 .. +7
  return f.v;
}

// ---------------- K0: prep (convert weights/codebook to f16, norms, zero hist)
__global__ __launch_bounds__(256)
void k_prep(const float* __restrict__ w1, const float* __restrict__ emb,
            const float* __restrict__ w2,
            _Float16* __restrict__ w1h, _Float16* __restrict__ eh,
            _Float16* __restrict__ w2h, float* __restrict__ enorm,
            int* __restrict__ hist) {
  const int t = blockIdx.x * 256 + threadIdx.x;
  if (t < 16384)  w1h[t] = (_Float16)w1[t];                 // [d][c], c contiguous
  if (t < 131072) eh[t]  = (_Float16)emb[t];                // [k][d], d contiguous
  if (t < 147456) {                                         // [tap][oc][ic]
    const int tap = t >> 14, rem = t & 16383;
    const int oc = rem >> 7, ic = rem & 127;
    w2h[t] = (_Float16)w2[oc * 1152 + ic * 9 + tap];
  }
  if (t < KCODE) {
    const float* e = emb + (size_t)t * DLAT;
    float s = 0.f;
    for (int d = 0; d < DLAT; ++d) s += e[d] * e[d];
    enorm[t] = s;
    hist[t] = 0;
  }
}

// ---------------- K1: 1x1 conv as WMMA GEMM -> x (fp32 + f16, BHWC) ----------
__global__ __launch_bounds__(256)
void k_conv1(const float* __restrict__ in, const _Float16* __restrict__ w1h,
             const float* __restrict__ b1, float* __restrict__ xf,
             _Float16* __restrict__ xh) {
  __shared__ __align__(16) _Float16 xT[128 * 136];   // [p][c], padded rows (16B aligned)
  const int tid = threadIdx.x;
  const int p0  = blockIdx.x * 128;                  // 128 positions, one image
  const int b   = p0 >> 12;
  const int hw0 = p0 & 4095;
  const float* src = in + (size_t)b * CIMG + hw0;

  // stage NCHW [c][p] tile -> LDS transposed f16 [p][c]
  for (int it = 0; it < 16; ++it) {
    const int idx = it * 256 + tid;
    const int c = idx >> 5, p = (idx & 31) * 4;
    float4 v = *(const float4*)(src + (size_t)c * HW + p);
    xT[(p + 0) * 136 + c] = (_Float16)v.x;
    xT[(p + 1) * 136 + c] = (_Float16)v.y;
    xT[(p + 2) * 136 + c] = (_Float16)v.z;
    xT[(p + 3) * 136 + c] = (_Float16)v.w;
  }
  __syncthreads();

  const int wave = tid >> 5, lane = tid & 31;
  const int nlane = lane & 15, khalf = lane >> 4;

  v16h af[4];
  #pragma unroll
  for (int kk = 0; kk < 4; ++kk)
    af[kk] = load_frag_A(xT, wave * 16, 136, kk * 32, lane);

  #pragma unroll
  for (int nt = 0; nt < 8; ++nt) {
    v8f acc = {};
    #pragma unroll
    for (int kk = 0; kk < 4; ++kk)
      acc = wmma_f16(af[kk], load_frag_B(w1h, nt * 16, 128, kk * 32, lane), acc);
    const float bias = b1[nt * 16 + nlane];
    #pragma unroll
    for (int r = 0; r < 8; ++r) {                    // D elem: m=r+8*khalf, n=nlane
      const int pos = p0 + wave * 16 + r + khalf * 8;
      const float v = acc[r] + bias;
      xf[(size_t)pos * DLAT + nt * 16 + nlane] = v;            // coalesced
      xh[(size_t)pos * DLAT + nt * 16 + nlane] = (_Float16)v;
    }
  }
}

// ---------------- K2: distances + argmin (16 pos x 1024 codes per wave) ------
__global__ __launch_bounds__(256)
void k_dist(const _Float16* __restrict__ xh, const _Float16* __restrict__ eh,
            const float* __restrict__ enorm, int* __restrict__ idx) {
  const int tid = threadIdx.x, wave = tid >> 5, lane = tid & 31;
  const int p0 = (blockIdx.x * 8 + wave) * 16;
  const int nlane = lane & 15, khalf = lane >> 4;

  v16h af[4];
  #pragma unroll
  for (int kk = 0; kk < 4; ++kk)
    af[kk] = load_frag_A(xh, p0, DLAT, kk * 32, lane);

  float minv[8]; int mini[8];
  #pragma unroll
  for (int r = 0; r < 8; ++r) { minv[r] = 3.4e38f; mini[r] = 0; }

  for (int ct = 0; ct < 64; ++ct) {                  // code tiles of 16
    const int c0 = ct * 16;
    v8f acc = {};
    #pragma unroll
    for (int kk = 0; kk < 4; ++kk)
      acc = wmma_f16(af[kk], load_frag_B(eh, c0, DLAT, kk * 32, lane), acc);
    const float ev = enorm[c0 + nlane];              // ||e||^2 (||x||^2 dropped: argmin-invariant)
    const int code = c0 + nlane;
    #pragma unroll
    for (int r = 0; r < 8; ++r) {
      const float d = ev - 2.0f * acc[r];
      if (d < minv[r]) { minv[r] = d; mini[r] = code; }
    }
  }
  // reduce across the 16 lanes of each half (same m, different code column)
  #pragma unroll
  for (int off = 1; off < 16; off <<= 1) {
    #pragma unroll
    for (int r = 0; r < 8; ++r) {
      const float ov = __shfl_xor(minv[r], off, 32);
      const int   oi = __shfl_xor(mini[r], off, 32);
      if (ov < minv[r] || (ov == minv[r] && oi < mini[r])) { minv[r] = ov; mini[r] = oi; }
    }
  }
  if (nlane == 0) {
    #pragma unroll
    for (int r = 0; r < 8; ++r)
      idx[p0 + r + khalf * 8] = mini[r];
  }
}

// ---------------- K3: gather quant, loss partials, histogram, q_nchw out -----
__global__ __launch_bounds__(256)
void k_quant(const int* __restrict__ idx, const float* __restrict__ emb,
             const float* __restrict__ xf, float* __restrict__ out,
             _Float16* __restrict__ qh, int* __restrict__ hist,
             float* __restrict__ partial) {
  const int p = blockIdx.x * 256 + threadIdx.x;     // one position per thread
  const int i = idx[p];
  const int b = p >> 12, hw = p & 4095;
  atomicAdd(&hist[i], 1);
  out[OUT_IDX + p] = (float)i;

  const float4* e4 = (const float4*)(emb + (size_t)i * DLAT);
  const float4* x4 = (const float4*)(xf + (size_t)p * DLAT);
  float* qo = out + OUT_Q + (size_t)b * CIMG + hw;
  float acc = 0.f;
  for (int d4 = 0; d4 < 32; ++d4) {
    const float4 e = e4[d4], x = x4[d4];
    const float dx = e.x - x.x, dy = e.y - x.y, dz = e.z - x.z, dw = e.w - x.w;
    acc += dx * dx + dy * dy + dz * dz + dw * dw;
    const int d = d4 * 4;
    qo[(size_t)(d + 0) * HW] = e.x;                 // NCHW, coalesced across lanes
    qo[(size_t)(d + 1) * HW] = e.y;
    qo[(size_t)(d + 2) * HW] = e.z;
    qo[(size_t)(d + 3) * HW] = e.w;
    union { _Float16 h[4]; uint2 u; } qq;
    qq.h[0] = (_Float16)e.x; qq.h[1] = (_Float16)e.y;
    qq.h[2] = (_Float16)e.z; qq.h[3] = (_Float16)e.w;
    *(uint2*)(qh + (size_t)p * DLAT + d) = qq.u;    // BHWC f16 for conv2
  }
  __shared__ float red[256];
  red[threadIdx.x] = acc;
  __syncthreads();
  for (int s = 128; s > 0; s >>= 1) {
    if (threadIdx.x < s) red[threadIdx.x] += red[threadIdx.x + s];
    __syncthreads();
  }
  if (threadIdx.x == 0) partial[blockIdx.x] = red[0];   // deterministic partials
}

// ---------------- K4: 3x3 conv as implicit WMMA GEMM -------------------------
__global__ __launch_bounds__(256)
void k_conv2(const _Float16* __restrict__ qh, const _Float16* __restrict__ w2h,
             const float* __restrict__ b2, float* __restrict__ out) {
  __shared__ __align__(16) float tbuf[8][16 * 20];  // per-wave transpose buffer
  const int tid = threadIdx.x, wave = tid >> 5, lane = tid & 31;
  const int wgid = blockIdx.x * 8 + wave;
  const int pg  = (wgid >> 1) * 32;                 // 32 positions (same row: 32|64)
  const int oc0 = (wgid & 1) * 64;                  // 64 output channels
  const int nlane = lane & 15, khalf = lane >> 4;
  const int b = pg >> 12;
  const int h = (pg & 4095) >> 6;

  v8f acc[4][2];
  const v8f vz = {};
  #pragma unroll
  for (int nt = 0; nt < 4; ++nt)
    for (int mt = 0; mt < 2; ++mt) acc[nt][mt] = vz;

  int posm[2], wm[2];
  #pragma unroll
  for (int mt = 0; mt < 2; ++mt) {
    posm[mt] = pg + mt * 16 + nlane;                // A-frag row m = lane&15
    wm[mt] = posm[mt] & 63;
  }

  for (int tap = 0; tap < 9; ++tap) {
    const int kh = tap / 3, kw = tap % 3;
    const int dh = kh - 1, dw = kw - 1;
    const int hp = h + dh;
    const bool rowok = (hp >= 0) && (hp < 64);
    const _Float16* wbase = w2h + (size_t)tap * 16384;
    #pragma unroll
    for (int kc = 0; kc < 4; ++kc) {
      v16h afr[2];
      #pragma unroll
      for (int mt = 0; mt < 2; ++mt) {
        const int wp = wm[mt] + dw;
        const bool ok = rowok && (wp >= 0) && (wp < 64);
        ABfrag f;
        f.q[0] = make_uint4(0u, 0u, 0u, 0u);
        f.q[1] = make_uint4(0u, 0u, 0u, 0u);
        if (ok) {                                   // zero-padding at borders
          const _Float16* p = qh + (size_t)(posm[mt] + dh * 64 + dw) * DLAT
                              + kc * 32 + khalf * 8;
          f.q[0] = *(const uint4*)(p);
          f.q[1] = *(const uint4*)(p + 16);
        }
        afr[mt] = f.v;
      }
      #pragma unroll
      for (int nt = 0; nt < 4; ++nt) {
        const v16h bf = load_frag_B(wbase, oc0 + nt * 16, 128, kc * 32, lane);
        acc[nt][0] = wmma_f16(afr[0], bf, acc[nt][0]);
        acc[nt][1] = wmma_f16(afr[1], bf, acc[nt][1]);
      }
    }
  }

  // epilogue: bias + LDS transpose so NCHW stores are contiguous b128s
  #pragma unroll
  for (int nt = 0; nt < 4; ++nt) {
    const float bias = b2[oc0 + nt * 16 + nlane];
    #pragma unroll
    for (int mt = 0; mt < 2; ++mt) {
      __syncthreads();                              // uniform across all waves
      #pragma unroll
      for (int r = 0; r < 8; ++r)
        tbuf[wave][nlane * 20 + r + khalf * 8] = acc[nt][mt][r] + bias;
      __syncthreads();
      const int nrow = lane >> 1, mo = (lane & 1) * 8;
      const float4 v0 = *(const float4*)&tbuf[wave][nrow * 20 + mo];
      const float4 v1 = *(const float4*)&tbuf[wave][nrow * 20 + mo + 4];
      const size_t dst = (size_t)OUT_NEW + (size_t)b * CIMG
                       + (size_t)(oc0 + nt * 16 + nrow) * HW
                       + (size_t)(pg + mt * 16 + mo);
      *(float4*)(out + dst)     = v0;
      *(float4*)(out + dst + 4) = v1;
    }
  }
}

// ---------------- K5: finalize loss + perplexity -----------------------------
__global__ __launch_bounds__(256)
void k_final(const int* __restrict__ hist, const float* __restrict__ partial,
             float* __restrict__ out) {
  __shared__ float red[256];
  const int tid = threadIdx.x;

  // sum of squared diffs (deterministic fixed-order reduction)
  red[tid] = partial[tid];
  __syncthreads();
  for (int s = 128; s > 0; s >>= 1) {
    if (tid < s) red[tid] += red[tid + s];
    __syncthreads();
  }
  const float sumsq = red[0];
  __syncthreads();

  // entropy of code usage
  float s = 0.f;
  for (int k = tid; k < KCODE; k += 256) {
    const float pk = (float)hist[k] * (1.0f / 65536.0f);
    s += pk * logf(pk + 1e-10f);
  }
  red[tid] = s;
  __syncthreads();
  for (int st = 128; st > 0; st >>= 1) {
    if (tid < st) red[tid] += red[tid + st];
    __syncthreads();
  }
  if (tid == 0) {
    out[OUT_PERP] = expf(-red[0]);
    // loss = (COMMIT + 1) * mean((quant - inp)^2), COMMIT = 0.25
    out[OUT_LOSS] = 1.25f * sumsq / 8388608.0f;
  }
}

// ---------------- launcher ---------------------------------------------------
extern "C" void kernel_launch(void* const* d_in, const int* in_sizes, int n_in,
                              void* d_out, int out_size, void* d_ws, size_t ws_size,
                              hipStream_t stream) {
  const float* in  = (const float*)d_in[0];   // [16,128,64,64]
  const float* w1  = (const float*)d_in[1];   // [128,128,1,1]
  const float* b1  = (const float*)d_in[2];   // [128]
  const float* emb = (const float*)d_in[3];   // [1024,128]
  const float* w2  = (const float*)d_in[4];   // [128,128,3,3]
  const float* b2  = (const float*)d_in[5];   // [128]
  float* out = (float*)d_out;

  char* ws = (char*)d_ws;
  float*    xf      = (float*)   (ws + 0);          // 33,554,432 B : x fp32 [pos][d]
  _Float16* xh      = (_Float16*)(ws + 33554432);   // 16,777,216 B : x f16
  _Float16* qh      = (_Float16*)(ws + 50331648);   // 16,777,216 B : quant f16
  _Float16* eh      = (_Float16*)(ws + 67108864);   //    262,144 B : emb f16
  _Float16* w1h     = (_Float16*)(ws + 67371008);   //     32,768 B
  _Float16* w2h     = (_Float16*)(ws + 67403776);   //    294,912 B : [tap][oc][ic]
  float*    enorm   = (float*)   (ws + 67698688);   //      4,096 B
  int*      idx     = (int*)     (ws + 67702784);   //    262,144 B
  int*      hist    = (int*)     (ws + 67964928);   //      4,096 B
  float*    partial = (float*)   (ws + 67969024);   //      1,024 B

  k_prep <<<576, 256, 0, stream>>>(w1, emb, w2, w1h, eh, w2h, enorm, hist);
  k_conv1<<<512, 256, 0, stream>>>(in, w1h, b1, xf, xh);
  k_dist <<<512, 256, 0, stream>>>(xh, eh, enorm, idx);
  k_quant<<<256, 256, 0, stream>>>(idx, emb, xf, out, qh, hist, partial);
  k_conv2<<<512, 256, 0, stream>>>(qh, w2h, b2, out);
  k_final<<<1,   256, 0, stream>>>(hist, partial, out);
}
